// ConvAct_69861938037159
// MI455X (gfx1250) — matrix operand order repaired and checked by reading, not compiled, not run on hardware
//
#include <hip/hip_runtime.h>
#include <hip/hip_bf16.h>

typedef __attribute__((ext_vector_type(16))) __bf16       v16bf;
typedef __attribute__((ext_vector_type(8)))  float        v8f;
typedef __attribute__((ext_vector_type(8)))  unsigned int v8u;
typedef __attribute__((ext_vector_type(4)))  unsigned int v4u;

#define HW    256
#define HW2   (HW * HW)
#define CIN   32
#define COUT  32
#define KTAP  9
#define NCOL  258   // image row + 2 halo columns (the conv zero padding)
#define CSTR  20    // words per staged column (16 channel-pairs + pad, 80B = 5*16B)

// hi = top 16 bits of f32 (exact bf16 truncation); lo = truncate(x - hi).
// Dropped lo*lo term ~2^-16 relative.
__device__ __forceinline__ void split2(float f0, float f1, unsigned& hi, unsigned& lo) {
  unsigned b0 = __float_as_uint(f0), b1 = __float_as_uint(f1);
  unsigned h0 = b0 & 0xffff0000u,    h1 = b1 & 0xffff0000u;
  hi = h1 | (b0 >> 16);
  float l0 = f0 - __uint_as_float(h0);
  float l1 = f1 - __uint_as_float(h1);
  lo = (__float_as_uint(l1) & 0xffff0000u) | (__float_as_uint(l0) >> 16);
}

__device__ __forceinline__ v8u lds_load8(const unsigned* p) {  // 16B-aligned
  v4u a = *(const v4u*)p;
  v4u b = *(const v4u*)(p + 4);
  v8u r;
  r[0] = a[0]; r[1] = a[1]; r[2] = a[2]; r[3] = a[3];
  r[4] = b[0]; r[5] = b[1]; r[6] = b[2]; r[7] = b[3];
  return r;
}

__device__ __forceinline__ v8f wmma_bf16(v8u a, v8u b, v8f c) {
  return __builtin_amdgcn_wmma_f32_16x16x32_bf16(
      false, __builtin_bit_cast(v16bf, a),
      false, __builtin_bit_cast(v16bf, b),
      (short)0, c, false, false);
}

__global__ __launch_bounds__(256)
void conv3x3_prelu_wmma(const float* __restrict__ x,
                        const float* __restrict__ weight,
                        const float* __restrict__ bias,
                        const float* __restrict__ prelu_a,
                        float* __restrict__ out) {
  // Weights pre-split hi/lo, packed in this kernel's contiguous-k A-fragment order:
  // word (((tap*2 + mtile)*32 + lane)*8 + v); K-pair(v,half) = 16h+2v
  __shared__ unsigned a_hi_lds[KTAP * 2 * 32 * 8];   // 18 KB
  __shared__ unsigned a_lo_lds[KTAP * 2 * 32 * 8];   // 18 KB
  // One staged image row (current kh), split hi/lo, packed channel pairs:
  // word [lc * CSTR + cp], lc = w+1 in [0,257], cp = channel pair 0..15.
  __shared__ unsigned xrow_hi[NCOL * CSTR];          // 20.2 KB
  __shared__ unsigned xrow_lo[NCOL * CSTR];          // 20.2 KB
  __shared__ float    bias_lds[COUT];

  for (int idx = threadIdx.x; idx < KTAP * 2 * 32 * 8; idx += 256) {
    int v    = idx & 7;
    int ln   = (idx >> 3) & 31;
    int tile = (idx >> 8) & 1;
    int tap  = idx >> 9;
    int hlf  = ln >> 4;
    int m    = tile * 16 + (ln & 15);
    int K0   = 16 * hlf + 2 * v;
    float f0 = weight[m * (CIN * 9) + (K0 + 0) * 9 + tap];
    float f1 = weight[m * (CIN * 9) + (K0 + 1) * 9 + tap];
    split2(f0, f1, a_hi_lds[idx], a_lo_lds[idx]);
  }
  if (threadIdx.x < COUT) bias_lds[threadIdx.x] = bias[threadIdx.x];
  // Halo columns (image cols -1 and 256) are the conv zero padding: zero once.
  if (threadIdx.x < 32) {
    int lc = (threadIdx.x >> 4) ? (NCOL - 1) : 0;
    int cp = threadIdx.x & 15;
    xrow_hi[lc * CSTR + cp] = 0u;
    xrow_lo[lc * CSTR + cp] = 0u;
  }

  const int lane  = threadIdx.x & 31;
  const int wv    = threadIdx.x >> 5;   // 8 waves per block
  const int half  = lane >> 4;
  const int col   = lane & 15;
  const int n     = blockIdx.x >> 8;
  const int h     = blockIdx.x & (HW - 1);
  const int wbase = wv * 32;            // wave tile: 32(M) x 32(N)
  const int w0    = wbase + col;        // N-tile 0 column
  const int w1    = w0 + 16;            // N-tile 1 column

  const float* xn = x + (size_t)n * CIN * HW2;

  v8f acc00 = {}; v8f acc01 = {}; v8f acc10 = {}; v8f acc11 = {};

  #pragma unroll
  for (int kh = 0; kh < 3; ++kh) {
    const int hm = h + kh - 1;
    if ((unsigned)hm >= (unsigned)HW) continue;   // block-uniform scalar branch

    // ---- Stage row hm: split once, reuse for all 3 kw taps ----
    __syncthreads();   // previous iteration's readers are done
    {
      const float* prow = xn + hm * HW + threadIdx.x;  // w = tid, coalesced
      #pragma unroll
      for (int cp = 0; cp < 16; ++cp) {
        float f0 = prow[(2 * cp + 0) * HW2];
        float f1 = prow[(2 * cp + 1) * HW2];
        unsigned hiw, low;
        split2(f0, f1, hiw, low);
        xrow_hi[(threadIdx.x + 1) * CSTR + cp] = hiw;
        xrow_lo[(threadIdx.x + 1) * CSTR + cp] = low;
      }
    }
    __syncthreads();

    #pragma unroll
    for (int kw = 0; kw < 3; ++kw) {
      const int tap = kh * 3 + kw;

      // A fragments: hi/lo x two M-tiles (32B-aligned, 2x ds_load_b128 each)
      const int abase = ((tap * 2) * 32 + lane) * 8;
      const v8u ah0 = lds_load8(&a_hi_lds[abase]);
      const v8u ah1 = lds_load8(&a_hi_lds[abase + 256]);
      const v8u al0 = lds_load8(&a_lo_lds[abase]);
      const v8u al1 = lds_load8(&a_lo_lds[abase + 256]);

      // B fragments straight from the staged row: 8 consecutive words each
      const int b0base = (w0 + kw) * CSTR + 8 * half;
      const int b1base = (w1 + kw) * CSTR + 8 * half;
      const v8u bh0 = lds_load8(&xrow_hi[b0base]);
      const v8u bl0 = lds_load8(&xrow_lo[b0base]);
      const v8u bh1 = lds_load8(&xrow_hi[b1base]);
      const v8u bl1 = lds_load8(&xrow_lo[b1base]);

      // D += Ahi*Bhi + Ahi*Blo + Alo*Bhi  (12 bf16 K=32 WMMAs per tap)
      acc00 = wmma_bf16(ah0, bh0, acc00);
      acc00 = wmma_bf16(ah0, bl0, acc00);
      acc00 = wmma_bf16(al0, bh0, acc00);

      acc10 = wmma_bf16(ah1, bh0, acc10);
      acc10 = wmma_bf16(ah1, bl0, acc10);
      acc10 = wmma_bf16(al1, bh0, acc10);

      acc01 = wmma_bf16(ah0, bh1, acc01);
      acc01 = wmma_bf16(ah0, bl1, acc01);
      acc01 = wmma_bf16(al0, bh1, acc01);

      acc11 = wmma_bf16(ah1, bh1, acc11);
      acc11 = wmma_bf16(ah1, bl1, acc11);
      acc11 = wmma_bf16(al1, bh1, acc11);
    }
  }

  const float pa = prelu_a[0];
  float* outn = out + (size_t)n * COUT * HW2;

  // C/D layout: VGPR r -> M = r + 8*half (within 16-row tile), N = lane&15
  #pragma unroll
  for (int r = 0; r < 8; ++r) {
    const int mlo = 8 * half + r;
    {
      const int m = mlo;  // M-tile 0
      float y0 = acc00[r] + bias_lds[m];
      y0 = y0 > 0.0f ? y0 : pa * y0;
      outn[((size_t)m * HW + h) * HW + (wbase + col)] = y0;
      float y1 = acc01[r] + bias_lds[m];
      y1 = y1 > 0.0f ? y1 : pa * y1;
      outn[((size_t)m * HW + h) * HW + (wbase + 16 + col)] = y1;
    }
    {
      const int m = 16 + mlo;  // M-tile 1
      float y0 = acc10[r] + bias_lds[m];
      y0 = y0 > 0.0f ? y0 : pa * y0;
      outn[((size_t)m * HW + h) * HW + (wbase + col)] = y0;
      float y1 = acc11[r] + bias_lds[m];
      y1 = y1 > 0.0f ? y1 : pa * y1;
      outn[((size_t)m * HW + h) * HW + (wbase + 16 + col)] = y1;
    }
  }
}

extern "C" void kernel_launch(void* const* d_in, const int* in_sizes, int n_in,
                              void* d_out, int out_size, void* d_ws, size_t ws_size,
                              hipStream_t stream) {
  (void)in_sizes; (void)n_in; (void)out_size; (void)d_ws; (void)ws_size;
  const float* x      = (const float*)d_in[0];
  const float* weight = (const float*)d_in[1];
  const float* bias   = (const float*)d_in[2];
  const float* pa     = (const float*)d_in[3];
  float* out          = (float*)d_out;

  dim3 grid(16 * 256);   // one block per (n, h) image row
  dim3 block(256);       // 8 wave32 -> full 32-channel x 256-column row
  conv3x3_prelu_wmma<<<grid, block, 0, stream>>>(x, weight, bias, pa, out);
}
